// SocialPooling_91285234909540
// MI455X (gfx1250) — matrix operand order repaired and checked by reading, not compiled, not run on hardware
//
#include <hip/hip_runtime.h>
#include <hip/hip_bf16.h>

// SocialPooling for MI455X (gfx1250, wave32).
//   features:    (8, 8192, 64) f32     coordinates: (8192, 2) f32
//   out:         (8, 8192, 64) f32
// Phase 1: per-row top-32 neighbor selection + normalization (one wave per row).
// Phase 2: gather-GEMM via v_wmma_f32_16x16x32_bf16 (f32 accumulate).
// Neighbor indices are stored as BYTE offsets (j * 256) so phase-2 gather
// addressing is a single 32-bit add folded into saddr+voffset global loads.

#define N_PTS    8192
#define N_B      8
#define N_D      64
#define NK       32      // MAX_NEIGHBORS
#define CAND_MAX 448     // expected ~257 in-radius candidates; +12 sigma headroom
#define ROW_B    (N_D * 4)   // feature row stride in bytes = 256

typedef __attribute__((ext_vector_type(16))) __bf16 v16bf;
typedef __attribute__((ext_vector_type(8)))  float  v8f;

__device__ __forceinline__ unsigned short f2bf(float f) {
  // round-to-nearest-even float -> bf16 bit pattern
  unsigned u = __float_as_uint(f);
  u += 0x7FFFu + ((u >> 16) & 1u);
  return (unsigned short)(u >> 16);
}

// ---------------------------------------------------------------------------
// Kernel 1: one wave32 per row. Build top-32 normalized neighbor list.
// ---------------------------------------------------------------------------
__global__ __launch_bounds__(32)
void sp_topk_kernel(const float* __restrict__ coords,
                    unsigned* __restrict__ nbr_off, float* __restrict__ wgt) {
  __shared__ float lw[CAND_MAX];
  __shared__ int   lj[CAND_MAX];
  __shared__ float selw[NK];
  __shared__ int   selj[NK];

  const int row  = blockIdx.x;
  const int lane = threadIdx.x;             // 0..31 (wave32)
  const float2 my = ((const float2*)coords)[row];

  // ---- scan all points, ballot-compact in-radius candidates into LDS ----
  int cnt = 0;                              // wave-uniform running count
  for (int t = lane; t < N_PTS; t += 32) {
    float2 p = ((const float2*)coords)[t];
    float dx = my.x - p.x, dy = my.y - p.y;
    float d2 = dx * dx + dy * dy;
    bool keep = (d2 <= 100.0f);             // radius 10 => d2 <= 100
    unsigned long long m = __ballot(keep);  // low 32 bits valid on wave32
    if (keep) {
      int pos = cnt + __popcll(m & ((1ull << lane) - 1ull));
      if (pos < CAND_MAX) {
        lw[pos] = expf(-0.02f * d2);        // exp(-d2 / (2*sigma^2)), sigma=5
        lj[pos] = t;
      }
    }
    cnt += __popcll(m);
  }
  __syncthreads();                          // 1-wave WG: barrier == S_NOP + waits

  // ---- 32 rounds of wave-parallel argmax == top-32 selection ----
  const int mcnt = (cnt < CAND_MAX) ? cnt : CAND_MAX;
  float ssum = 0.0f;
  int   nsel = 0;
  for (int it = 0; it < NK; ++it) {
    float bv = 0.0f; int bs = -1;
    for (int p = lane; p < mcnt; p += 32) {
      float wv = lw[p];
      if (wv > bv) { bv = wv; bs = p; }
    }
    #pragma unroll
    for (int off = 16; off > 0; off >>= 1) {
      float ov = __shfl_xor(bv, off, 32);
      int   os = __shfl_xor(bs, off, 32);
      if (ov > bv) { bv = ov; bs = os; }
    }
    if (bs < 0) break;                      // uniform: fewer than 32 candidates
    ssum += bv;
    if (lane == 0) { selw[it] = bv; selj[it] = lj[bs]; lw[bs] = -1.0f; }
    ++nsel;
    __syncthreads();                        // order the mark before next scan
  }

  const float inv = 1.0f / fmaxf(ssum, 1e-8f);
  __syncthreads();
  float w = 0.0f; unsigned joff = 0;
  if (lane < nsel) { w = selw[lane] * inv; joff = (unsigned)selj[lane] * ROW_B; }
  nbr_off[row * NK + lane] = joff;          // padded slots: joff=0, w=0 (no-op)
  wgt[row * NK + lane]     = w;
}

// ---------------------------------------------------------------------------
// Kernel 2: 16 output rows per WG, 4 waves = 4 d-blocks of 16.
// Per tile row c: out[d, c] += A(16d x 32k gathered feats) * w_c via WMMA.
// ---------------------------------------------------------------------------
__global__ __launch_bounds__(128)
void sp_pool_kernel(const float* __restrict__ feat,
                    const unsigned* __restrict__ nbr_off,
                    const float* __restrict__ wgt,
                    float* __restrict__ out) {
  __shared__ unsigned s_jo[16][NK];   // neighbor byte offsets (j * 256)
  __shared__ float    s_w[16][NK];    // normalized weights

  const int tile = blockIdx.x;
  const int tid  = threadIdx.x;
  for (int p = tid; p < 16 * NK; p += 128) {
    s_jo[p >> 5][p & 31] = nbr_off[tile * 16 * NK + p];
    s_w [p >> 5][p & 31] = wgt    [tile * 16 * NK + p];
  }
  __syncthreads();

  const int lane  = tid & 31;
  const int wave  = tid >> 5;
  const int dbase = wave << 4;                   // d-block: 0,16,32,48
  const int mrow  = lane & 15;                   // A's M (=d offset) and D's N
  const int half  = lane >> 4;
  const unsigned dByte = (unsigned)(dbase + mrow) * 4u;  // per-lane byte offset

  for (int b = 0; b < N_B; ++b) {
    const char* fb = (const char*)(feat + (size_t)b * N_PTS * N_D); // SGPR base
    v8f acc = {};
    for (int c = 0; c < 16; ++c) {
      // A fragment: 16x32 bf16, A[m][k] = feat[b][ j(c,k) ][ dbase+m ]
      // layout (ISA 7.12.2, 16-bit A 16x32): lane<16 -> K {0..7,16..23},
      // lane>=16 -> K {8..15,24..31}; M = lane%16.
      union { v16bf v; unsigned short us[16]; } av;
      #pragma unroll
      for (int e = 0; e < 16; ++e) {
        int k = (e & 7) + ((e >> 3) << 4) + (half << 3);
        unsigned off = s_jo[c][k] + dByte;       // one 32-bit v_add per gather
        float f = *(const float*)(fb + off);     // saddr + u32 voffset load
        av.us[e] = f2bf(f);
      }
      // B fragment: 32x16 bf16, only column n==c nonzero (row-c weights).
      // layout (matches A): lane L -> n = L%16,
      // elements e -> K = (e&7) + 16*(e/8) + 8*(L/16).
      union { v16bf v; unsigned short us[16]; } bv;
      const bool on = (mrow == c);
      #pragma unroll
      for (int e = 0; e < 16; ++e) {
        int k = (e & 7) + ((e >> 3) << 4) + (half << 3);
        bv.us[e] = on ? f2bf(s_w[c][k]) : (unsigned short)0;
      }
      acc = __builtin_amdgcn_wmma_f32_16x16x32_bf16(
          /*neg_a=*/false, av.v, /*neg_b=*/false, bv.v,
          /*c_mod=*/(short)0, acc, /*reuse_a=*/false, /*reuse_b=*/false);
    }
    // D layout: lane L: n = L%16; VGPR r holds M = r + 8*(L/16).
    // => each lane owns 8 consecutive d-floats: 32B coalesced store.
    // Output is written once and never re-read: non-temporal to spare L2
    // capacity for the feature-gather working set.
    const int i = tile * 16 + mrow;
    float* op = out + ((size_t)b * N_PTS + i) * N_D + dbase + (half << 3);
    #pragma unroll
    for (int r = 0; r < 8; ++r) __builtin_nontemporal_store(acc[r], op + r);
  }
}

// ---------------------------------------------------------------------------
extern "C" void kernel_launch(void* const* d_in, const int* in_sizes, int n_in,
                              void* d_out, int out_size, void* d_ws, size_t ws_size,
                              hipStream_t stream) {
  const float* feat   = (const float*)d_in[0];   // (8, 8192, 64) f32
  const float* coords = (const float*)d_in[1];   // (8192, 2) f32

  // workspace: 8192*32 u32 offsets + 8192*32 f32 weights = 2 MB
  unsigned* nbr_off = (unsigned*)d_ws;
  float*    wgt     = (float*)((char*)d_ws + sizeof(unsigned) * (size_t)N_PTS * NK);

  sp_topk_kernel<<<N_PTS, 32, 0, stream>>>(coords, nbr_off, wgt);
  sp_pool_kernel<<<N_PTS / 16, 128, 0, stream>>>(feat, nbr_off, wgt, (float*)d_out);
}
